// MoEGate_43963285242624
// MI455X (gfx1250) — matrix operand order repaired and checked by reading, not compiled, not run on hardware
//
#include <hip/hip_runtime.h>
#include <hip/hip_bf16.h>

typedef __attribute__((ext_vector_type(16))) __bf16 v16bf;
typedef __attribute__((ext_vector_type(4)))  __bf16 v4bf;
typedef __attribute__((ext_vector_type(8)))  float  v8f;
typedef __attribute__((ext_vector_type(4)))  float  v4f;

static constexpr int H    = 4096;  // hidden dim (K)
static constexpr int E    = 256;   // experts (N)
static constexpr int TOPK = 8;
static constexpr int NGRP = 8;     // 8 groups x 32 experts

// ---------------------------------------------------------------------------
// Prep kernel: decompose W (fp32, [E,H]) into bf16 hi/lo planes once.
// ~8 MB of HBM traffic, negligible vs the 268 MB x stream; W planes stay
// L2-resident afterwards.
// ---------------------------------------------------------------------------
__global__ __launch_bounds__(256) void split_w_kernel(
    const float* __restrict__ w, v4bf* __restrict__ whi,
    v4bf* __restrict__ wlo) {
  const size_t i = (size_t)blockIdx.x * 256 + threadIdx.x;  // v4f index
  v4f v = reinterpret_cast<const v4f*>(w)[i];
  v4bf h, l;
#pragma unroll
  for (int j = 0; j < 4; ++j) {
    __bf16 hb = (__bf16)v[j];                // RNE fp32->bf16
    h[j] = hb;
    l[j] = (__bf16)(v[j] - (float)hb);       // residual (next 8 mantissa bits)
  }
  whi[i] = h;
  wlo[i] = l;
}

// ---------------------------------------------------------------------------
// A-fragment: load 16 fp32 from an x row, split to bf16 hi/lo in registers.
// A (16x32, 16-bit): lane<16 -> row=lane,    K = kb+[0..15]
//                    lane>=16 -> row=lane-16, K = kb+16+[0..15]
// hi loop, residual-float loop, lo loop: adjacent conversions can fuse into
// v_cvt_pk_bf16_f32 pairs and land in contiguous WMMA operand VGPRs.
// ---------------------------------------------------------------------------
__device__ inline void splitA(const float* __restrict__ rowp, int kb, int lane,
                              v16bf& hi, v16bf& lo) {
  const int koff = (lane >> 4) * 16;
  const v4f* p0 = reinterpret_cast<const v4f*>(rowp + kb + koff);
  const v4f* p1 = reinterpret_cast<const v4f*>(rowp + kb + 8 + koff);
  v4f f0 = p0[0], f1 = p0[1], f2 = p1[0], f3 = p1[1];
  float f[16] = {f0[0], f0[1], f0[2], f0[3], f1[0], f1[1], f1[2], f1[3],
                 f2[0], f2[1], f2[2], f2[3], f3[0], f3[1], f3[2], f3[3]};
#pragma unroll
  for (int i = 0; i < 16; ++i) hi[i] = (__bf16)f[i];
  float r[16];
#pragma unroll
  for (int i = 0; i < 16; ++i) r[i] = f[i] - (float)hi[i];
#pragma unroll
  for (int i = 0; i < 16; ++i) lo[i] = (__bf16)r[i];
}

// ---------------------------------------------------------------------------
// B-fragment: pre-split bf16, loaded directly in the same contiguous k order as A.
// B (32x16, 16-bit): lane<16 -> col=lane,    K = kb+[0..15]
//                    lane>=16 -> col=lane-16, K = kb+16+[0..15]
// -> one aligned 32-byte vector load (2x global_load_b128), zero VALU.
// ---------------------------------------------------------------------------
__device__ inline v16bf loadBpre(const __bf16* __restrict__ colp, int kb,
                                 int lane) {
  const int koff = (lane >> 4) * 16;
  return *reinterpret_cast<const v16bf*>(colp + kb + koff);
}

// ---------------------------------------------------------------------------
// Kernel 1: scores[m, e] = sigmoid(x[m,:] . W[e,:])  via split-bf16 WMMA.
// Block: 128 threads = 4 waves as 2(M) x 2(N). Block tile: 64 tokens x 256
// experts (x read from HBM exactly once). Wave tile: 32 x 128 = 2x8 accum
// tiles -> 48 WMMAs per K-chunk of 32 against a fixed ~90 VALU of A-split
// (3 WMMAs per tile: Ah*Bh + Ah*Bl + Al*Bh; the Al*Bl term is ~2^-16).
// ---------------------------------------------------------------------------
__global__ __launch_bounds__(128) void gate_gemm_kernel(
    const float* __restrict__ x, const __bf16* __restrict__ whi,
    const __bf16* __restrict__ wlo, float* __restrict__ scores) {
  const int lane  = threadIdx.x & 31;
  const int wave  = threadIdx.x >> 5;
  const int waveM = wave & 1;   // 0..1
  const int waveN = wave >> 1;  // 0..1
  const int rowBase = blockIdx.x * 64 + waveM * 32;
  const int colBase = waveN * 128;
  const int lrow = lane & 15;

  const float* xr[2];
  xr[0] = x + (size_t)(rowBase + lrow) * H;
  xr[1] = x + (size_t)(rowBase + 16 + lrow) * H;
  const __bf16* bhr[8];
  const __bf16* blr[8];
#pragma unroll
  for (int n = 0; n < 8; ++n) {
    const size_t col = (size_t)(colBase + n * 16 + lrow);
    bhr[n] = whi + col * H;
    blr[n] = wlo + col * H;
  }

  v8f acc[2][8];
#pragma unroll
  for (int m = 0; m < 2; ++m)
#pragma unroll
    for (int n = 0; n < 8; ++n)
#pragma unroll
      for (int i = 0; i < 8; ++i) acc[m][n][i] = 0.0f;

  for (int kb = 0; kb < H; kb += 32) {
    v16bf ah[2], al[2], bh[8], bl[8];
#pragma unroll
    for (int n = 0; n < 8; ++n) {
      bh[n] = loadBpre(bhr[n], kb, lane);
      bl[n] = loadBpre(blr[n], kb, lane);
    }
#pragma unroll
    for (int m = 0; m < 2; ++m) splitA(xr[m], kb, lane, ah[m], al[m]);

#pragma unroll
    for (int m = 0; m < 2; ++m)
#pragma unroll
      for (int n = 0; n < 8; ++n) {
        acc[m][n] = __builtin_amdgcn_wmma_f32_16x16x32_bf16(
            false, ah[m], false, bh[n], (short)0, acc[m][n], false, false);
        acc[m][n] = __builtin_amdgcn_wmma_f32_16x16x32_bf16(
            false, ah[m], false, bl[n], (short)0, acc[m][n], false, false);
        acc[m][n] = __builtin_amdgcn_wmma_f32_16x16x32_bf16(
            false, al[m], false, bh[n], (short)0, acc[m][n], false, false);
      }
  }

  // Epilogue: sigmoid, store. C/D layout: elem i at lane L -> row = i + 8*(L>>4),
  // col = L&15 within the 16x16 tile.
  const int rOff = 8 * (lane >> 4);
  const int c    = lane & 15;
#pragma unroll
  for (int m = 0; m < 2; ++m)
#pragma unroll
    for (int n = 0; n < 8; ++n)
#pragma unroll
      for (int i = 0; i < 8; ++i) {
        float g = acc[m][n][i];
        float s = 1.0f / (1.0f + __expf(-g));
        scores[(size_t)(rowBase + m * 16 + rOff + i) * E +
               (colBase + n * 16 + c)] = s;
      }
}

// ---------------------------------------------------------------------------
// Kernel 2: per-token gating. One wave32 per token; lane owns experts
// [lane*8, lane*8+8) so each group of 32 experts == 4 consecutive lanes.
// ---------------------------------------------------------------------------
__global__ __launch_bounds__(256) void gate_topk_kernel(
    const float* __restrict__ scores, const float* __restrict__ bias,
    int* __restrict__ out_idx, float* __restrict__ out_val) {
  const int lane  = threadIdx.x & 31;
  const int wave  = threadIdx.x >> 5;
  const int token = blockIdx.x * 8 + wave;

  const float* srow = scores + (size_t)token * E;
  const v4f* sp = reinterpret_cast<const v4f*>(srow + lane * 8);
  const v4f* bp = reinterpret_cast<const v4f*>(bias + lane * 8);
  v4f s0 = sp[0], s1 = sp[1];
  v4f b0 = bp[0], b1 = bp[1];

  float sv[8], bv[8];
#pragma unroll
  for (int i = 0; i < 4; ++i) {
    sv[i] = s0[i];         sv[i + 4] = s1[i];
    bv[i] = s0[i] + b0[i]; bv[i + 4] = s1[i] + b1[i];
  }

  // ---- group scores: sum of top-2 biased scores within each group of 32 ----
  float t1 = -1e30f, t2 = -1e30f;
#pragma unroll
  for (int i = 0; i < 8; ++i) {
    if (bv[i] > t1) { t2 = t1; t1 = bv[i]; }
    else if (bv[i] > t2) { t2 = bv[i]; }
  }
#pragma unroll
  for (int off = 1; off <= 2; off <<= 1) {  // merge top-2 across 4 lanes
    float o1 = __shfl_xor(t1, off, 32);
    float o2 = __shfl_xor(t2, off, 32);
    float m1 = fmaxf(t1, o1);
    float m2 = fmaxf(fminf(t1, o1), fmaxf(t2, o2));
    t1 = m1; t2 = m2;
  }
  const float gscore = t1 + t2;

  float gs[NGRP];
#pragma unroll
  for (int j = 0; j < NGRP; ++j) gs[j] = __shfl(gscore, j * 4, 32);
  const int g = lane >> 2;
  int above = 0;
#pragma unroll
  for (int j = 0; j < NGRP; ++j)
    above += (gs[j] > gs[g]) || (gs[j] == gs[g] && j > g);
  const bool keep = (above < 4);  // keep top-4 groups; ties drop lower index
  if (!keep) {
#pragma unroll
    for (int i = 0; i < 8; ++i) bv[i] = 0.0f;  // reference masks with 0.0
  }

  // ---- top-8 over masked biased scores; carry unbiased score along ----
  int myIdx = 0; float myOrig = 0.0f; float sum = 0.0f;
#pragma unroll
  for (int r = 0; r < TOPK; ++r) {
    float bestv = -1e30f, bestO = 0.0f; int besti = 0;
#pragma unroll
    for (int i = 0; i < 8; ++i)
      if (bv[i] > bestv) { bestv = bv[i]; bestO = sv[i]; besti = i; }
    int bestIdx = lane * 8 + besti;
#pragma unroll
    for (int off = 16; off >= 1; off >>= 1) {  // wave argmax, low idx wins ties
      float ov = __shfl_xor(bestv, off, 32);
      float oo = __shfl_xor(bestO, off, 32);
      int   oi = __shfl_xor(bestIdx, off, 32);
      if (ov > bestv || (ov == bestv && oi < bestIdx)) {
        bestv = ov; bestO = oo; bestIdx = oi;
      }
    }
    if (lane == r) { myIdx = bestIdx; myOrig = bestO; }
    sum += bestO;
    if (lane == (bestIdx >> 3)) {  // knock out the winner
      const int slot = bestIdx & 7;
#pragma unroll
      for (int i = 0; i < 8; ++i)
        if (i == slot) bv[i] = -1e30f;
    }
  }

  if (lane < TOPK) {
    out_idx[(size_t)token * TOPK + lane] = myIdx;
    out_val[(size_t)token * TOPK + lane] =
        myOrig / (sum + 1e-20f) * 2.5f;  // norm_topk_prob + routed scaling
  }
}

// ---------------------------------------------------------------------------
extern "C" void kernel_launch(void* const* d_in, const int* in_sizes, int n_in,
                              void* d_out, int out_size, void* d_ws,
                              size_t ws_size, hipStream_t stream) {
  const float* x    = (const float*)d_in[0];  // [B*S, H] fp32
  const float* w    = (const float*)d_in[1];  // [E, H]   fp32
  const float* bias = (const float*)d_in[2];  // [E]      fp32

  const int M = in_sizes[0] / H;  // B*S = 16384 tokens

  // workspace layout: scores [M*E] fp32 | W_hi [E*H] bf16 | W_lo [E*H] bf16
  float*  scores = (float*)d_ws;
  __bf16* whi    = (__bf16*)((char*)d_ws + (size_t)M * E * sizeof(float));
  __bf16* wlo    = whi + (size_t)E * H;

  int*   out_idx = (int*)d_out;                       // M*TOPK int32 indices
  float* out_val = (float*)d_out + (size_t)M * TOPK;  // M*TOPK fp32 weights

  split_w_kernel<<<(E * H / 4) / 256, 256, 0, stream>>>(
      w, (v4bf*)whi, (v4bf*)wlo);
  gate_gemm_kernel<<<M / 64, 128, 0, stream>>>(x, whi, wlo, scores);
  gate_topk_kernel<<<M / 8, 256, 0, stream>>>(scores, bias, out_idx, out_val);
  (void)n_in; (void)out_size; (void)ws_size;
}
